// RelativeMultiHeadAttention_33913061769758
// MI455X (gfx1250) — matrix operand order repaired and checked by reading, not compiled, not run on hardware
//
#include <hip/hip_runtime.h>

#define HEADS    16
#define BATCH    2048
#define SEQ      32
#define MROWS    (BATCH * SEQ)          // 65536
#define ATTN_SCALE 0.125f               // 64^-0.5

typedef __attribute__((ext_vector_type(16))) __bf16 v16bf;
typedef __attribute__((ext_vector_type(8)))  float  v8f;

union FragBF {
    v16bf v;
    uint4 q[2];
    unsigned short s[16];
};
union AccF {
    v8f v;
    float f[8];
};

__device__ __forceinline__ unsigned short f2bf(float f) {
    unsigned int u = __float_as_uint(f);
    u += 0x7FFFu + ((u >> 16) & 1u);          // round-to-nearest-even
    return (unsigned short)(u >> 16);
}
__device__ __forceinline__ uint2 pack4(float4 f) {
    uint2 r;
    r.x = (unsigned)f2bf(f.x) | ((unsigned)f2bf(f.y) << 16);
    r.y = (unsigned)f2bf(f.z) | ((unsigned)f2bf(f.w) << 16);
    return r;
}

// CDNA5 async memory->LDS copy (ASYNCcnt-tracked), GVS addressing:
// dst VGPR = LDS byte address, vaddr = 32-bit byte offset, saddr = 64-bit base.
__device__ __forceinline__ void async_ld16(unsigned lds_addr, const void* base,
                                           unsigned byte_off) {
    asm volatile("global_load_async_to_lds_b128 %0, %1, %2"
                 :: "v"(lds_addr), "v"(byte_off), "s"(base)
                 : "memory");
}
__device__ __forceinline__ void wait_async0() {
    asm volatile("s_wait_asynccnt 0x0" ::: "memory");
}

// ---------------------------------------------------------------------------
// fp32 -> bf16 streaming convert (vector of 4 per thread)
// ---------------------------------------------------------------------------
__global__ __launch_bounds__(256) void cvt_f32_bf16(
    const float4* __restrict__ in, uint2* __restrict__ out, int n4)
{
    int i = blockIdx.x * blockDim.x + threadIdx.x;
    if (i < n4) out[i] = pack4(in[i]);
}

// ---------------------------------------------------------------------------
// C[M,Nc] = A[M,K] @ W[Nc,K]^T + bias ; A,W bf16 ; out bf16 or fp32.
// Block 256 thr = 8 waves ; tile 128x128 ; staged K-step 64 (2 WMMA k-slices).
// Double-buffered LDS fed by global_load_async_to_lds_b128, one barrier/tile.
// ---------------------------------------------------------------------------
template<bool OUT_BF16>
__global__ __launch_bounds__(256) void gemm_bf16_wmma(
    const unsigned short* __restrict__ A, const unsigned short* __restrict__ W,
    const float* __restrict__ bias, void* __restrict__ Out,
    int Ncols, int K)
{
    __shared__ unsigned short sA[2][128 * 64];   // 2 x 16 KB
    __shared__ unsigned short sB[2][128 * 64];   // 2 x 16 KB

    const int tid  = threadIdx.x;
    const int lane = tid & 31;
    const int lo   = lane & 15;
    const int hi   = lane >> 4;
    const int wave = tid >> 5;
    const int wm   = wave & 3;
    const int wn   = wave >> 2;
    const long mBase = (long)blockIdx.y * 128;
    const long nBase = (long)blockIdx.x * 128;

    v8f acc[2][4];
#pragma unroll
    for (int mt = 0; mt < 2; ++mt)
#pragma unroll
        for (int nt = 0; nt < 4; ++nt) acc[mt][nt] = (v8f)(0.0f);

    // issue async copies of K-slice kt into buffer buf (4 x 16B per matrix/thread)
    auto issue = [&](int kt, int buf) {
        const unsigned aLds = (unsigned)(size_t)(&sA[buf][0]);
        const unsigned bLds = (unsigned)(size_t)(&sB[buf][0]);
        const int kOff = kt * 64;
#pragma unroll
        for (int i = 0; i < 4; ++i) {
            int c = tid + i * 256;                 // 1024 chunks of 8 bf16
            int row = c >> 3, k8 = (c & 7) * 8;
            unsigned l = (unsigned)(row * 64 + k8) * 2u;
            async_ld16(aLds + l, A,
                       (unsigned)(((mBase + row) * (long)K + kOff + k8) * 2));
            async_ld16(bLds + l, W,
                       (unsigned)(((nBase + row) * (long)K + kOff + k8) * 2));
        }
    };

    const int KT = K >> 6;                         // K / 64
    issue(0, 0);

    for (int kt = 0; kt < KT; ++kt) {
        const int buf = kt & 1;
        wait_async0();                             // my writes into sX[buf] done
        __syncthreads();                           // everyone's done (+ reads of buf^1)
        if (kt + 1 < KT) issue(kt + 1, buf ^ 1);   // overlap with MMA below

#pragma unroll
        for (int ki = 0; ki < 2; ++ki) {
            const int k0 = ki * 32;
            FragBF af[2], bfr[4];
#pragma unroll
            for (int mt = 0; mt < 2; ++mt) {
                int row = wm * 32 + mt * 16 + lo;
                int kh  = hi ? 8 : 0;              // A: K = [kh..+7, kh+16..+23]
                af[mt].q[0] = *(const uint4*)&sA[buf][row * 64 + k0 + kh];
                af[mt].q[1] = *(const uint4*)&sA[buf][row * 64 + k0 + kh + 16];
            }
#pragma unroll
            for (int nt = 0; nt < 4; ++nt) {
                int row = wn * 64 + nt * 16 + lo;
                int kh  = hi ? 8 : 0;              // B: K = [kh..+7, kh+16..+23]
                bfr[nt].q[0] = *(const uint4*)&sB[buf][row * 64 + k0 + kh];
                bfr[nt].q[1] = *(const uint4*)&sB[buf][row * 64 + k0 + kh + 16];
            }
#pragma unroll
            for (int mt = 0; mt < 2; ++mt)
#pragma unroll
                for (int nt = 0; nt < 4; ++nt)
                    acc[mt][nt] = __builtin_amdgcn_wmma_f32_16x16x32_bf16(
                        false, af[mt].v, false, bfr[nt].v,
                        (short)0, acc[mt][nt], false, false);
        }
    }

    // epilogue: + bias, store
#pragma unroll
    for (int mt = 0; mt < 2; ++mt) {
#pragma unroll
        for (int nt = 0; nt < 4; ++nt) {
            long col = nBase + wn * 64 + nt * 16 + lo;
            float bv = bias[col];
            AccF u; u.v = acc[mt][nt];
#pragma unroll
            for (int r = 0; r < 8; ++r) {
                long row = mBase + wm * 32 + mt * 16 + r + hi * 8;
                float val = u.f[r] + bv;
                if (OUT_BF16)
                    ((unsigned short*)Out)[row * (long)Ncols + col] = f2bf(val);
                else
                    ((float*)Out)[row * (long)Ncols + col] = val;
            }
        }
    }
}

// ---------------------------------------------------------------------------
// Attention: one wave per (b,h). qkv bf16 [65536][3072], col = 3idx*1024+h*64+d.
// S = q k^T * scale + rel_bias ; softmax rows ; O = P V -> attOut bf16
// ---------------------------------------------------------------------------
__global__ __launch_bounds__(256) void attn_wmma(
    const unsigned short* __restrict__ qkv,
    const float* __restrict__ rel_bias,
    unsigned short* __restrict__ attOut)
{
    __shared__ unsigned short sP[8][32 * 32];   // per-wave probs, bf16

    const int tid  = threadIdx.x;
    const int lane = tid & 31;
    const int lo   = lane & 15;
    const int hi   = lane >> 4;
    const int wave = tid >> 5;
    const int pair = blockIdx.x * 8 + wave;     // (b,h)
    const int b    = pair >> 4;
    const int h    = pair & 15;
    const long rowBase = (long)b * SEQ;

    // ---- S = q @ k^T ----
    v8f s[2][2];
#pragma unroll
    for (int mt = 0; mt < 2; ++mt)
#pragma unroll
        for (int nt = 0; nt < 2; ++nt) s[mt][nt] = (v8f)(0.0f);

#pragma unroll
    for (int ks = 0; ks < 2; ++ks) {            // head_dim 64 = 2 K-steps
        FragBF qa[2], kb[2];
#pragma unroll
        for (int mt = 0; mt < 2; ++mt) {
            long r = rowBase + mt * 16 + lo;
            const unsigned short* p =
                qkv + r * 3072 + h * 64 + ks * 32 + (hi ? 8 : 0);
            qa[mt].q[0] = *(const uint4*)p;
            qa[mt].q[1] = *(const uint4*)(p + 16);
        }
#pragma unroll
        for (int nt = 0; nt < 2; ++nt) {
            long r = rowBase + nt * 16 + lo;
            const unsigned short* p =
                qkv + r * 3072 + 1024 + h * 64 + ks * 32 + (hi ? 8 : 0);
            kb[nt].q[0] = *(const uint4*)p;
            kb[nt].q[1] = *(const uint4*)(p + 16);
        }
#pragma unroll
        for (int mt = 0; mt < 2; ++mt)
#pragma unroll
            for (int nt = 0; nt < 2; ++nt)
                s[mt][nt] = __builtin_amdgcn_wmma_f32_16x16x32_bf16(
                    false, qa[mt].v, false, kb[nt].v,
                    (short)0, s[mt][nt], false, false);
    }

    // ---- scale + rel_bias + softmax over keys (C-layout columns) ----
    float sc[2][2][8];
#pragma unroll
    for (int mt = 0; mt < 2; ++mt)
#pragma unroll
        for (int nt = 0; nt < 2; ++nt) {
            AccF u; u.v = s[mt][nt];
#pragma unroll
            for (int r = 0; r < 8; ++r) {
                int M  = mt * 16 + r + hi * 8;       // query index
                int Nc = nt * 16 + lo;               // key index
                sc[mt][nt][r] = u.f[r] * ATTN_SCALE +
                                rel_bias[h * 1024 + M * 32 + Nc];
            }
        }

#pragma unroll
    for (int mt = 0; mt < 2; ++mt) {
#pragma unroll
        for (int r = 0; r < 8; ++r) {
            float m = fmaxf(sc[mt][0][r], sc[mt][1][r]);
#pragma unroll
            for (int off = 1; off <= 8; off <<= 1)   // 16-lane row group
                m = fmaxf(m, __shfl_xor(m, off, 32));
            float e0 = __expf(sc[mt][0][r] - m);
            float e1 = __expf(sc[mt][1][r] - m);
            float sum = e0 + e1;
#pragma unroll
            for (int off = 1; off <= 8; off <<= 1)
                sum += __shfl_xor(sum, off, 32);
            float inv = __frcp_rn(sum);
            int M = mt * 16 + r + hi * 8;
            sP[wave][M * 32 + lo]      = f2bf(e0 * inv);
            sP[wave][M * 32 + 16 + lo] = f2bf(e1 * inv);
        }
    }
    __syncthreads();   // order cross-lane LDS probs before A-layout reload

    // ---- O = P @ V ----
    FragBF pa[2];
#pragma unroll
    for (int mt = 0; mt < 2; ++mt) {
        int row = mt * 16 + lo;
        int kh  = hi ? 8 : 0;
        pa[mt].q[0] = *(const uint4*)&sP[wave][row * 32 + kh];
        pa[mt].q[1] = *(const uint4*)&sP[wave][row * 32 + kh + 16];
    }

    v8f o[2][4];
#pragma unroll
    for (int mt = 0; mt < 2; ++mt)
#pragma unroll
        for (int nt = 0; nt < 4; ++nt) o[mt][nt] = (v8f)(0.0f);

#pragma unroll
    for (int nt = 0; nt < 4; ++nt) {            // head-dim tiles
        FragBF vb;
        const long vcol = 2048 + h * 64 + nt * 16 + lo;
        const int  kh   = hi ? 8 : 0;           // key rows: [kh..+7, kh+16..+23]
#pragma unroll
        for (int j = 0; j < 16; ++j)
            vb.s[j] = qkv[(rowBase + kh + (j < 8 ? j : j + 8)) * 3072 + vcol];
#pragma unroll
        for (int mt = 0; mt < 2; ++mt)
            o[mt][nt] = __builtin_amdgcn_wmma_f32_16x16x32_bf16(
                false, pa[mt].v, false, vb.v,
                (short)0, o[mt][nt], false, false);
    }

    // ---- store attn output bf16: attOut[b*32+n][h*64+d] ----
#pragma unroll
    for (int mt = 0; mt < 2; ++mt)
#pragma unroll
        for (int nt = 0; nt < 4; ++nt) {
            AccF u; u.v = o[mt][nt];
            long col = h * 64 + nt * 16 + lo;
#pragma unroll
            for (int r = 0; r < 8; ++r) {
                long row = rowBase + mt * 16 + r + hi * 8;
                attOut[row * 1024 + col] = f2bf(u.f[r]);
            }
        }
}

// ---------------------------------------------------------------------------
extern "C" void kernel_launch(void* const* d_in, const int* in_sizes, int n_in,
                              void* d_out, int out_size, void* d_ws, size_t ws_size,
                              hipStream_t stream) {
    (void)in_sizes; (void)n_in; (void)out_size; (void)ws_size;
    const float* x      = (const float*)d_in[0];   // [65536,1024]
    const float* W_qkv  = (const float*)d_in[1];   // [3072,1024]
    const float* b_qkv  = (const float*)d_in[2];   // [3072]
    const float* W_proj = (const float*)d_in[3];   // [1024,1024]
    const float* b_proj = (const float*)d_in[4];   // [1024]
    const float* relb   = (const float*)d_in[5];   // [16,32,32]

    unsigned short* qkv = (unsigned short*)d_ws;            // 65536*3072 bf16
    unsigned short* att = qkv + (size_t)MROWS * 3072;       // 65536*1024 bf16
    unsigned short* xbf = att + (size_t)MROWS * 1024;       // 65536*1024 bf16
    unsigned short* wqb = xbf + (size_t)MROWS * 1024;       // 3072*1024 bf16
    unsigned short* wpb = wqb + (size_t)3072 * 1024;        // 1024*1024 bf16
    float* out = (float*)d_out;

    dim3 blk(256);
    // 0) one-time fp32 -> bf16 converts (streaming, ~12 us at HBM rate)
    cvt_f32_bf16<<<(MROWS * 1024 / 4) / 256, blk, 0, stream>>>(
        (const float4*)x, (uint2*)xbf, MROWS * 1024 / 4);
    cvt_f32_bf16<<<(3072 * 1024 / 4) / 256, blk, 0, stream>>>(
        (const float4*)W_qkv, (uint2*)wqb, 3072 * 1024 / 4);
    cvt_f32_bf16<<<(1024 * 1024 / 4) / 256, blk, 0, stream>>>(
        (const float4*)W_proj, (uint2*)wpb, 1024 * 1024 / 4);

    // 1) qkv = x @ W_qkv^T + b_qkv   (bf16 out)
    gemm_bf16_wmma<true><<<dim3(3072 / 128, MROWS / 128), blk, 0, stream>>>(
        xbf, wqb, b_qkv, qkv, 3072, 1024);
    // 2) attention per (b,h)
    attn_wmma<<<(BATCH * HEADS) / 8, blk, 0, stream>>>(qkv, relb, att);
    // 3) out = att @ W_proj^T + b_proj   (fp32 out)
    gemm_bf16_wmma<false><<<dim3(1024 / 128, MROWS / 128), blk, 0, stream>>>(
        att, wpb, b_proj, out, 1024, 1024);
}
